// Model_33517924778303
// MI455X (gfx1250) — compile-verified, matrix operand order checked by reading, not run on hardware
//
#include <hip/hip_runtime.h>
#include <stdint.h>
#include <stddef.h>

typedef _Float16 half_t;
typedef __attribute__((ext_vector_type(16))) _Float16 v16h;
typedef __attribute__((ext_vector_type(8)))  _Float16 v8h;
typedef __attribute__((ext_vector_type(8)))  float    v8f;

#define TPB 256

// =====================================================================
// WMMA GEMM (fast path): C(MxN) = A(MxK) @ B(KxN), f32 storage, f16 math.
// Preconditions (host dispatch): M%16==0, N%16==0, K%32==0.
// Block = 8 waves; each wave computes ONE 16-row M-tile x FOUR 16-col
// N-tiles (16x64 of C), reusing a single A fragment across 4 WMMAs.
// The 32x64 B k-slab is cooperatively staged into LDS (transposed,
// 48-half column stride -> 16B-aligned ds_load_b128 per fragment half).
// Tail N-tiles: staging addresses clamped, stores guarded (wave-uniform).
// Fragment packing per CDNA5 ISA 7.12.2 (wave32):
//   A 16x32 f16: lane<16 -> row=lane&15, kb=0 ; lane>=16 -> kb=8
//                halves j<8 : K=kb+j ; j>=8 : K=kb+16+(j-8)
//   B 32x16 f16: col=lane&15, same K pattern
//   C 16x16 f32: VGPR r -> row = r + (lane<16?0:8), col = lane&15
// =====================================================================
__global__ void __launch_bounds__(256)
k_gemm_wmma_fast(const float* __restrict__ A, const float* __restrict__ B,
                 float* __restrict__ C, int M, int N, int K)
{
  __shared__ _Float16 ldsB[64 * 48];               // [col][k], stride 48 halves (96B)
  const int tid  = (int)threadIdx.x;
  const int lane = tid & 31;
  const int w    = tid >> 5;                        // wave in block (0..7)
  const int ntM  = M >> 4;
  const int ntN  = N >> 4;
  int tm = (int)blockIdx.x * 8 + w;
  if (tm >= ntM) tm = ntM - 1;                      // clamp: keeps barriers valid
  const int tn4   = (int)blockIdx.y;                // group of 4 N-tiles
  const int cbase = tn4 * 64;
  const int arow  = tm * 16 + (lane & 15);
  const int col15 = lane & 15;
  const int kb    = (lane < 16) ? 0 : 8;
  const float* Arow = A + (size_t)arow * K;

  v8f acc0 = {}, acc1 = {}, acc2 = {}, acc3 = {};
  for (int k0 = 0; k0 < K; k0 += 32) {
    // ---- stage B slab (32 rows x 64 cols) into LDS, transposed, clamped ----
#pragma unroll
    for (int it = 0; it < 8; ++it) {                // 2048 elems / 256 threads
      const int e = tid + it * 256;
      const int k = e >> 6, c = e & 63;
      int gc = cbase + c; gc = (gc < N) ? gc : (N - 1);   // branch-free clamp
      ldsB[c * 48 + k] = (half_t)B[(size_t)(k0 + k) * N + gc];
    }
    __syncthreads();

    // ---- A fragment: 8x float2, fully unconditional, shared by 4 WMMAs ----
    const float* Ap = Arow + k0 + kb;
    if (k0 + 32 < K) __builtin_prefetch(Ap + 32, 0, 1);   // global_prefetch_b8
    v16h a;
#pragma unroll
    for (int j2 = 0; j2 < 4; ++j2) {
      const float2 t = *(const float2*)(Ap + 2 * j2);
      a[2 * j2]     = (half_t)t.x;
      a[2 * j2 + 1] = (half_t)t.y;
      const float2 u = *(const float2*)(Ap + 16 + 2 * j2);
      a[8 + 2 * j2]     = (half_t)u.x;
      a[8 + 2 * j2 + 1] = (half_t)u.y;
    }

    // ---- 4 B fragments from LDS (2x ds_load_b128 each), 4 WMMAs ----
#pragma unroll
    for (int t = 0; t < 4; ++t) {
      const int cb = (t * 16 + col15) * 48;
      const v8h b0 = *(const v8h*)&ldsB[cb + kb];          // halves 0..7
      const v8h b1 = *(const v8h*)&ldsB[cb + kb + 16];     // halves 8..15
      v16h b;
#pragma unroll
      for (int j = 0; j < 8; ++j) { b[j] = b0[j]; b[8 + j] = b1[j]; }
      if      (t == 0) acc0 = __builtin_amdgcn_wmma_f32_16x16x32_f16(false, a, false, b, (short)0, acc0, false, false);
      else if (t == 1) acc1 = __builtin_amdgcn_wmma_f32_16x16x32_f16(false, a, false, b, (short)0, acc1, false, false);
      else if (t == 2) acc2 = __builtin_amdgcn_wmma_f32_16x16x32_f16(false, a, false, b, (short)0, acc2, false, false);
      else             acc3 = __builtin_amdgcn_wmma_f32_16x16x32_f16(false, a, false, b, (short)0, acc3, false, false);
    }
    __syncthreads();                                 // protect ldsB before next stage
  }

  const int rbase = tm * 16 + ((lane < 16) ? 0 : 8);
#pragma unroll
  for (int t = 0; t < 4; ++t) {
    const int tnt = tn4 * 4 + t;
    if (tnt < ntN) {                                 // wave-uniform guard
      const int ccol = tnt * 16 + col15;
      const v8f accv = (t == 0) ? acc0 : (t == 1) ? acc1 : (t == 2) ? acc2 : acc3;
#pragma unroll
      for (int r = 0; r < 8; ++r)
        C[(size_t)(rbase + r) * N + ccol] = accv[r];
    }
  }
}

// =====================================================================
// WMMA GEMM (generic path): ragged K and/or N (K=3,6,8,16,67,...; N=3,8).
// M%16==0 still guaranteed. Branch-free: addresses are clamped in-bounds and
// out-of-range lanes zeroed with VALU selects -> no exec-mask load branches.
// =====================================================================
__global__ void k_gemm_wmma_gen(const float* __restrict__ A, const float* __restrict__ B,
                                float* __restrict__ C, int M, int N, int K)
{
  const int lane = (int)(threadIdx.x & 31u);
  const int wave = (int)((blockIdx.x * blockDim.x + threadIdx.x) >> 5);
  const int ntN = (N + 15) >> 4;
  const int ntM = M >> 4;
  if (wave >= ntM * ntN) return;                  // wave-uniform exit
  const int tm = wave / ntN, tn = wave - tm * ntN;
  const int arow = tm * 16 + (lane & 15);
  const int bcolu = tn * 16 + (lane & 15);
  const int bcol  = (bcolu < N) ? bcolu : (N - 1);   // clamped address
  const bool cin  = (bcolu < N);
  const int kb = (lane < 16) ? 0 : 8;
  v8f acc = {};
  for (int k0 = 0; k0 < K; k0 += 32) {
    v16h a, b;
#pragma unroll
    for (int j = 0; j < 16; ++j) {
      const int kk  = k0 + kb + j + ((j < 8) ? 0 : 8);
      const int kkc = (kk < K) ? kk : (K - 1);
      float av = A[(size_t)arow * K + kkc];
      float bv = B[(size_t)kkc * N + bcol];
      av = (kk < K) ? av : 0.0f;                   // v_cndmask, no branch
      bv = (kk < K && cin) ? bv : 0.0f;
      a[j] = (half_t)av;
      b[j] = (half_t)bv;
    }
    acc = __builtin_amdgcn_wmma_f32_16x16x32_f16(false, a, false, b, (short)0, acc,
                                                 false, false);
  }
  const int rbase = tm * 16 + ((lane < 16) ? 0 : 8);
#pragma unroll
  for (int r = 0; r < 8; ++r)
    if (cin) C[(size_t)(rbase + r) * N + bcolu] = acc[r];
}

// ============ BatchNorm (training-mode, per-channel over rows, no affine) ====
// Deterministic: per-block channel-striped partial sums, single-block finalize.
__global__ void k_bn_part(const float* __restrict__ X, float* __restrict__ part,
                          int R, int C, int rpb)
{
  const int r0 = blockIdx.x * rpb;
  int r1 = r0 + rpb; if (r1 > R) r1 = R;
  for (int ch = threadIdx.x; ch < C; ch += blockDim.x) {
    float s = 0.f, q = 0.f;
    for (int r = r0; r < r1; ++r) { const float v = X[(size_t)r * C + ch]; s += v; q += v * v; }
    part[(size_t)blockIdx.x * 2 * C + ch]     = s;
    part[(size_t)blockIdx.x * 2 * C + C + ch] = q;
  }
}
__global__ void k_bn_fin(const float* __restrict__ part, float* __restrict__ stats,
                         int nb, int C)
{
  for (int ch = threadIdx.x; ch < 2 * C; ch += blockDim.x) {
    float s = 0.f;
    for (int b = 0; b < nb; ++b) s += part[(size_t)b * 2 * C + ch];
    stats[ch] = s;
  }
}
__global__ void k_bn_apply(const float* __restrict__ X, const float* __restrict__ stats,
                           const float* __restrict__ res, float* __restrict__ Y,
                           int R, int C, int relu)
{
  const size_t i = (size_t)blockIdx.x * blockDim.x + threadIdx.x;
  if (i >= (size_t)R * C) return;
  const int ch = (int)(i % (size_t)C);
  const float invR = 1.0f / (float)R;
  const float m = stats[ch] * invR;
  const float v = stats[C + ch] * invR - m * m;
  float y = (X[i] - m) * rsqrtf(v + 1e-5f);
  if (res) y += res[i];
  if (relu) y = fmaxf(y, 0.0f);
  Y[i] = y;
}

// ============ brute-force KNN, k=16, register insertion sort ============
__global__ void k_knn(const float* __restrict__ Q, const float* __restrict__ Rf,
                      int nq, int nr, int* __restrict__ out)
{
  const int i = blockIdx.x * blockDim.x + threadIdx.x;
  if (i >= nq) return;
  const float qx = Q[3*i], qy = Q[3*i+1], qz = Q[3*i+2];
  float bd[16]; int bi[16];
#pragma unroll
  for (int j = 0; j < 16; ++j) { bd[j] = 3.4e38f; bi[j] = 0; }
  for (int j = 0; j < nr; ++j) {
    const float dx = Rf[3*j] - qx, dy = Rf[3*j+1] - qy, dz = Rf[3*j+2] - qz;
    const float d = dx*dx + dy*dy + dz*dz;
    if (d < bd[15]) {
      int p = 15;
      while (p > 0 && bd[p-1] > d) { bd[p] = bd[p-1]; bi[p] = bi[p-1]; --p; }
      bd[p] = d; bi[p] = j;
    }
  }
#pragma unroll
  for (int j = 0; j < 16; ++j) out[(size_t)i * 16 + j] = bi[j];
}

// ============ farthest point sampling: single WG, dist array in LDS ============
__global__ void __launch_bounds__(1024)
k_fps(const float* __restrict__ P, int n, int m, int* __restrict__ out)
{
  __shared__ float dist[8192];
  __shared__ float rb[1024];
  __shared__ int   ra[1024];
  __shared__ int   s_last;
  const int tid = threadIdx.x;
  for (int i = tid; i < n; i += blockDim.x) dist[i] = 1e10f;
  if (tid == 0) { out[0] = 0; s_last = 0; }
  __syncthreads();
  int last = 0;
  for (int it = 1; it < m; ++it) {
    const float lx = P[3*last], ly = P[3*last+1], lz = P[3*last+2];
    float best = -1.0f; int barg = 0;
    for (int i = tid; i < n; i += blockDim.x) {
      const float dx = P[3*i] - lx, dy = P[3*i+1] - ly, dz = P[3*i+2] - lz;
      const float d = dx*dx + dy*dy + dz*dz;
      const float nd = fminf(dist[i], d);
      dist[i] = nd;
      if (nd > best) { best = nd; barg = i; }
    }
    rb[tid] = best; ra[tid] = barg;
    __syncthreads();
    for (int s = (int)blockDim.x >> 1; s > 0; s >>= 1) {
      if (tid < s) {
        if (rb[tid+s] > rb[tid] || (rb[tid+s] == rb[tid] && ra[tid+s] < ra[tid])) {
          rb[tid] = rb[tid+s]; ra[tid] = ra[tid+s];
        }
      }
      __syncthreads();
    }
    if (tid == 0) { s_last = ra[0]; out[it] = ra[0]; }
    __syncthreads();
    last = s_last;
  }
}

// ============ small elementwise / gather kernels ============
__global__ void k_concat2(const float* __restrict__ a, int ca, const float* __restrict__ b,
                          int cb, float* __restrict__ o, int n)
{
  const int ct = ca + cb;
  const size_t i = (size_t)blockIdx.x * blockDim.x + threadIdx.x;
  if (i >= (size_t)n * ct) return;
  const int row = (int)(i / ct), ch = (int)(i % ct);
  o[i] = (ch < ca) ? a[(size_t)row * ca + ch] : b[(size_t)row * cb + (ch - ca)];
}

__global__ void k_gather_pr(const float* __restrict__ p, const int* __restrict__ idx,
                            float* __restrict__ pr, int n)
{
  const size_t i = (size_t)blockIdx.x * blockDim.x + threadIdx.x;
  if (i >= (size_t)n * 48) return;
  const int pt = (int)(i / 48); const int rem = (int)(i % 48);
  const int ns = rem / 3, d = rem % 3;
  const int nb = idx[(size_t)pt * 16 + ns];
  pr[i] = p[(size_t)nb * 3 + d] - p[(size_t)pt * 3 + d];
}

__global__ void k_build_w(const float* __restrict__ Kf, const float* __restrict__ Qf,
                          const float* __restrict__ PE, const int* __restrict__ idx,
                          float* __restrict__ W, int n, int c)
{
  const size_t i = (size_t)blockIdx.x * blockDim.x + threadIdx.x;
  if (i >= (size_t)n * 16 * c) return;
  const int pt = (int)(i / ((size_t)16 * c));
  const int rem = (int)(i % ((size_t)16 * c));
  const int ns = rem / c, ch = rem % c;
  const int nb = idx[(size_t)pt * 16 + ns];
  W[i] = Kf[(size_t)nb * c + ch] - Qf[(size_t)pt * c + ch] + PE[i];
}

__global__ void k_build_xv(const float* __restrict__ V, const float* __restrict__ PE,
                           const int* __restrict__ idx, float* __restrict__ XV, int n, int c)
{
  const size_t i = (size_t)blockIdx.x * blockDim.x + threadIdx.x;
  if (i >= (size_t)n * 16 * c) return;
  const int pt = (int)(i / ((size_t)16 * c));
  const int rem = (int)(i % ((size_t)16 * c));
  const int ns = rem / c, ch = rem % c;
  const int nb = idx[(size_t)pt * 16 + ns];
  XV[i] = V[(size_t)nb * c + ch] + PE[i];
}

__global__ void k_softmax_ns(float* __restrict__ W, int n, int cs)
{
  const int i = blockIdx.x * blockDim.x + threadIdx.x;
  if (i >= n * cs) return;
  const int pt = i / cs, ch = i % cs;
  const size_t base = (size_t)pt * 16 * cs + ch;
  float mx = -3.4e38f;
#pragma unroll
  for (int j = 0; j < 16; ++j) mx = fmaxf(mx, W[base + (size_t)j * cs]);
  float e[16]; float s = 0.f;
#pragma unroll
  for (int j = 0; j < 16; ++j) { e[j] = __expf(W[base + (size_t)j * cs] - mx); s += e[j]; }
  const float inv = 1.0f / s;
#pragma unroll
  for (int j = 0; j < 16; ++j) W[base + (size_t)j * cs] = e[j] * inv;
}

__global__ void k_combine(const float* __restrict__ XV, const float* __restrict__ W,
                          float* __restrict__ out, int n, int c)
{
  const size_t i = (size_t)blockIdx.x * blockDim.x + threadIdx.x;
  if (i >= (size_t)n * c) return;
  const int pt = (int)(i / c), ch = (int)(i % c);
  const int cs = c >> 3;
  const int wc = ch & (cs - 1);       // cs is a power of two
  float s = 0.f;
#pragma unroll
  for (int j = 0; j < 16; ++j)
    s += XV[((size_t)pt * 16 + j) * c + ch] * W[((size_t)pt * 16 + j) * cs + wc];
  out[i] = s;
}

__global__ void k_gather_rows3(const float* __restrict__ p, const int* __restrict__ fidx,
                               float* __restrict__ o, int m)
{
  const int i = blockIdx.x * blockDim.x + threadIdx.x;
  if (i >= m * 3) return;
  const int r = i / 3, d = i % 3;
  o[i] = p[(size_t)fidx[r] * 3 + d];
}

__global__ void k_gather_g(const float* __restrict__ p, const float* __restrict__ x,
                           const int* __restrict__ nidx, const float* __restrict__ newp,
                           float* __restrict__ g, int m, int c)
{
  const int din = 3 + c;
  const size_t i = (size_t)blockIdx.x * blockDim.x + threadIdx.x;
  if (i >= (size_t)m * 16 * din) return;
  const int pt = (int)(i / ((size_t)16 * din));
  const int rem = (int)(i % ((size_t)16 * din));
  const int ns = rem / din, ch = rem % din;
  const int nb = nidx[(size_t)pt * 16 + ns];
  g[i] = (ch < 3) ? (p[(size_t)nb * 3 + ch] - newp[(size_t)pt * 3 + ch])
                  : x[(size_t)nb * c + (ch - 3)];
}

__global__ void k_maxpool_ns(const float* __restrict__ Y, float* __restrict__ o, int m, int c)
{
  const size_t i = (size_t)blockIdx.x * blockDim.x + threadIdx.x;
  if (i >= (size_t)m * c) return;
  const int pt = (int)(i / c), ch = (int)(i % c);
  float mx = -3.4e38f;
#pragma unroll
  for (int j = 0; j < 16; ++j) mx = fmaxf(mx, Y[((size_t)pt * 16 + j) * c + ch]);
  o[i] = mx;
}

__global__ void k_mean_rows(const float* __restrict__ X, float* __restrict__ o, int R, int C)
{
  const int ch = blockIdx.x * blockDim.x + threadIdx.x;
  if (ch >= C) return;
  float s = 0.f;
  for (int r = 0; r < R; ++r) s += X[(size_t)r * C + ch];
  o[ch] = s / (float)R;
}

__global__ void k_gemv_relu(const float* __restrict__ x, const float* __restrict__ W,
                            float* __restrict__ y, int K, int N)
{
  const int n = blockIdx.x * blockDim.x + threadIdx.x;
  if (n >= N) return;
  float s = 0.f;
  for (int k = 0; k < K; ++k) s += x[k] * W[(size_t)k * N + n];
  y[n] = fmaxf(s, 0.f);
}

__global__ void k_bcast_concat(const float* __restrict__ x, const float* __restrict__ g,
                               float* __restrict__ o, int n, int c)
{
  const size_t i = (size_t)blockIdx.x * blockDim.x + threadIdx.x;
  if (i >= (size_t)n * 2 * c) return;
  const int ch = (int)(i % (size_t)(2 * c));
  const int row = (int)(i / (size_t)(2 * c));
  o[i] = (ch < c) ? x[(size_t)row * c + ch] : g[ch - c];
}

__global__ void k_interp_add(const float* __restrict__ pc, const float* __restrict__ pf,
                             const float* __restrict__ xc, const float* __restrict__ a,
                             float* __restrict__ o, int nc, int nf, int c)
{
  const int i = blockIdx.x * blockDim.x + threadIdx.x;
  if (i >= nf) return;
  const float px = pf[3*i], py = pf[3*i+1], pz = pf[3*i+2];
  float d0 = 3.4e38f, d1 = 3.4e38f, d2 = 3.4e38f;
  int i0 = 0, i1 = 0, i2 = 0;
  for (int j = 0; j < nc; ++j) {
    const float dx = pc[3*j]-px, dy = pc[3*j+1]-py, dz = pc[3*j+2]-pz;
    const float d = dx*dx + dy*dy + dz*dz;
    if (d < d0)      { d2 = d1; i2 = i1; d1 = d0; i1 = i0; d0 = d; i0 = j; }
    else if (d < d1) { d2 = d1; i2 = i1; d1 = d;  i1 = j; }
    else if (d < d2) { d2 = d;  i2 = j; }
  }
  float w0 = 1.0f / (sqrtf(fmaxf(d0, 1e-12f)) + 1e-8f);
  float w1 = 1.0f / (sqrtf(fmaxf(d1, 1e-12f)) + 1e-8f);
  float w2 = 1.0f / (sqrtf(fmaxf(d2, 1e-12f)) + 1e-8f);
  const float inv = 1.0f / (w0 + w1 + w2);
  w0 *= inv; w1 *= inv; w2 *= inv;
  const size_t b0 = (size_t)i0 * c, b1 = (size_t)i1 * c, b2 = (size_t)i2 * c, ob = (size_t)i * c;
  for (int ch = 0; ch < c; ++ch)
    o[ob + ch] = a[ob + ch] + w0 * xc[b0 + ch] + w1 * xc[b1 + ch] + w2 * xc[b2 + ch];
}

// =====================================================================
// Host orchestration
// =====================================================================
extern "C" void kernel_launch(void* const* d_in, const int* in_sizes, int n_in,
                              void* d_out, int out_size, void* d_ws, size_t ws_size,
                              hipStream_t stream)
{
  (void)in_sizes; (void)n_in;
  static const int NP[5]   = {8192, 2048, 512, 128, 32};
  static const int CHN[5]  = {64, 128, 256, 512, 1024};
  static const int NBLK[5] = {2, 3, 4, 6, 3};

  // ---- parameter pytree walk (JAX leaf order: dicts by sorted key) ----
  // top-level insertion order: params, p0, f0; params: 'dec' < 'enc';
  // stage: 'blocks' < 'down'/'up'; block: lin1, lin3, tr{k,p1,p2,q,v,w1,w2}
  int li = 0;
  auto nxt = [&]() -> const float* { return (const float*)d_in[li++]; };
  struct TrP  { const float *q,*k,*v,*p1,*p2,*w1,*w2; };
  struct BlkP { const float *lin1,*lin3; TrP tr; };
  BlkP decBlk[5]; const float* decUp1[5]; const float* decUp2[5];
  BlkP encBlk[5][5]; const float* encDown[5];
  auto readBlk = [&](BlkP& b) {
    b.lin1 = nxt(); b.lin3 = nxt();
    b.tr.k = nxt(); b.tr.p1 = nxt(); b.tr.p2 = nxt();
    b.tr.q = nxt(); b.tr.v = nxt(); b.tr.w1 = nxt(); b.tr.w2 = nxt();
  };
  for (int i = 0; i < 5; ++i) { readBlk(decBlk[i]); decUp1[i] = nxt(); decUp2[i] = nxt(); }
  for (int i = 0; i < 5; ++i) {
    for (int b = 0; b < NBLK[i] - 1; ++b) readBlk(encBlk[i][b]);
    encDown[i] = nxt();
  }
  const float* p0 = nxt();
  const float* f0 = nxt();

  // ---- workspace bump allocator ----
  char* base = (char*)d_ws;
  size_t off = 0;
  auto walloc = [&](size_t bytes) -> void* {
    void* p = base + off;
    off += (bytes + 255) & ~(size_t)255;
    return p;
  };

  float* psb[5] = {nullptr, nullptr, nullptr, nullptr, nullptr};
  const float* psc[5]; psc[0] = p0;
  for (int i = 1; i < 5; ++i) { psb[i] = (float*)walloc((size_t)NP[i] * 3 * 4); psc[i] = psb[i]; }
  float* xs[5];   for (int i = 0; i < 5; ++i) xs[i]   = (float*)walloc((size_t)NP[i] * CHN[i] * 4);
  int*   idxs[5]; for (int i = 0; i < 5; ++i) idxs[i] = (int*)  walloc((size_t)NP[i] * 16 * 4);
  float* decx[5]; for (int i = 0; i < 5; ++i) decx[i] = (float*)walloc((size_t)NP[i] * CHN[i] * 4);
  float* stats = (float*)walloc((size_t)2 * 1024 * 4);
  float* part  = (float*)walloc((size_t)64 * 2 * 1024 * 4);
  float* meanb = (float*)walloc((size_t)1024 * 4);
  float* gvec  = (float*)walloc((size_t)1024 * 4);
  const size_t tmpMark = off;

  // stage-0 attention temp peak is ~92 MB; bail deterministically if ws too small
  if (tmpMark + ((size_t)128 << 20) > ws_size) {
    hipMemsetAsync(d_out, 0, (size_t)out_size * sizeof(float), stream);
    return;
  }
  auto treset = [&]() { off = tmpMark; };
  auto grid1  = [&](size_t cnt) -> unsigned { return (unsigned)((cnt + TPB - 1) / TPB); };

  // GEMM dispatch: fast LDS/WMMA 4-tile path when K%32==0 && N%16==0.
  auto gemm = [&](const float* A, const float* B, float* C, int M, int N, int K) {
    if ((K & 31) == 0 && (N & 15) == 0) {
      const int ntN4 = ((N >> 4) + 3) >> 2;              // groups of 4 N-tiles
      dim3 grid((unsigned)(((M >> 4) + 7) >> 3), (unsigned)ntN4);
      k_gemm_wmma_fast<<<grid, 256, 0, stream>>>(A, B, C, M, N, K);
    } else {
      const int waves = (M >> 4) * ((N + 15) >> 4);
      const int blocks = (waves + 7) >> 3;               // 8 waves per block
      k_gemm_wmma_gen<<<blocks, 256, 0, stream>>>(A, B, C, M, N, K);
    }
  };
  auto bn = [&](const float* X, float* Y, int R, int C, const float* res, int relu) {
    int nb = R < 64 ? R : 64; if (nb < 1) nb = 1;
    const int rpb = (R + nb - 1) / nb;
    k_bn_part<<<nb, TPB, 0, stream>>>(X, part, R, C, rpb);
    k_bn_fin<<<1, TPB, 0, stream>>>(part, stats, nb, C);
    k_bn_apply<<<grid1((size_t)R * C), TPB, 0, stream>>>(X, stats, res, Y, R, C, relu);
  };

  // ---- point-transformer vector attention layer ----
  auto pt_layer = [&](const TrP& tr, const float* p, const float* x, const int* idx,
                      int n, int c, float* out) {
    const int cs = c >> 3;
    float* q  = (float*)walloc((size_t)n * c * 4);
    float* kf = (float*)walloc((size_t)n * c * 4);
    float* v  = (float*)walloc((size_t)n * c * 4);
    gemm(x, tr.q, q,  n, c, c);
    gemm(x, tr.k, kf, n, c, c);
    gemm(x, tr.v, v,  n, c, c);
    float* pr = (float*)walloc((size_t)n * 16 * 3 * 4);
    k_gather_pr<<<grid1((size_t)n * 48), TPB, 0, stream>>>(p, idx, pr, n);
    float* t3 = (float*)walloc((size_t)n * 16 * 3 * 4);
    gemm(pr, tr.p1, t3, n * 16, 3, 3);
    bn(t3, t3, n * 16, 3, nullptr, 1);
    float* pe = (float*)walloc((size_t)n * 16 * c * 4);
    gemm(t3, tr.p2, pe, n * 16, c, 3);
    float* w = (float*)walloc((size_t)n * 16 * c * 4);
    k_build_w<<<grid1((size_t)n * 16 * c), TPB, 0, stream>>>(kf, q, pe, idx, w, n, c);
    bn(w, w, n * 16, c, nullptr, 1);
    float* w2 = (float*)walloc((size_t)n * 16 * cs * 4);
    gemm(w, tr.w1, w2, n * 16, cs, c);
    bn(w2, w2, n * 16, cs, nullptr, 1);
    float* w3 = (float*)walloc((size_t)n * 16 * cs * 4);
    gemm(w2, tr.w2, w3, n * 16, cs, cs);
    k_softmax_ns<<<grid1((size_t)n * cs), TPB, 0, stream>>>(w3, n, cs);
    // reuse w as xv = v[idx] + pe
    k_build_xv<<<grid1((size_t)n * 16 * c), TPB, 0, stream>>>(v, pe, idx, w, n, c);
    k_combine<<<grid1((size_t)n * c), TPB, 0, stream>>>(w, w3, out, n, c);
  };

  auto pt_block = [&](const BlkP& bp, const float* p, const float* xin, float* xout,
                      const int* idx, int n, int c) {
    treset();
    float* y1 = (float*)walloc((size_t)n * c * 4);
    gemm(xin, bp.lin1, y1, n, c, c);
    bn(y1, y1, n, c, nullptr, 1);
    float* att = (float*)walloc((size_t)n * c * 4);
    pt_layer(bp.tr, p, y1, idx, n, c, att);
    bn(att, att, n, c, nullptr, 1);
    float* y3 = (float*)walloc((size_t)n * c * 4);
    gemm(att, bp.lin3, y3, n, c, c);
    bn(y3, xout, n, c, xin, 1);          // xout = relu(xin + bn(y3))
  };

  // ===================== encoder =====================
  treset();
  float* x0in = (float*)walloc((size_t)8192 * 6 * 4);
  k_concat2<<<grid1((size_t)8192 * 6), TPB, 0, stream>>>(p0, 3, f0, 3, x0in, 8192);
  gemm(x0in, encDown[0], xs[0], 8192, 64, 6);
  bn(xs[0], xs[0], 8192, 64, nullptr, 1);
  k_knn<<<grid1(8192), TPB, 0, stream>>>(psc[0], psc[0], 8192, 8192, idxs[0]);
  for (int b = 0; b < NBLK[0] - 1; ++b)
    pt_block(encBlk[0][b], psc[0], xs[0], xs[0], idxs[0], 8192, 64);

  for (int i = 1; i < 5; ++i) {
    const int npre = NP[i-1], cpre = CHN[i-1];
    const int m = NP[i], c = CHN[i];
    const int din = 3 + cpre;
    treset();
    int* fidx = (int*)walloc((size_t)m * 4);
    k_fps<<<1, 1024, 0, stream>>>(psc[i-1], npre, m, fidx);
    k_gather_rows3<<<grid1((size_t)m * 3), TPB, 0, stream>>>(psc[i-1], fidx, psb[i], m);
    int* nidx = (int*)walloc((size_t)m * 16 * 4);
    k_knn<<<grid1(m), TPB, 0, stream>>>(psc[i], psc[i-1], m, npre, nidx);
    float* g = (float*)walloc((size_t)m * 16 * din * 4);
    k_gather_g<<<grid1((size_t)m * 16 * din), TPB, 0, stream>>>(psc[i-1], xs[i-1], nidx,
                                                               psc[i], g, m, cpre);
    float* yb = (float*)walloc((size_t)m * 16 * c * 4);
    gemm(g, encDown[i], yb, m * 16, c, din);
    bn(yb, yb, m * 16, c, nullptr, 1);
    k_maxpool_ns<<<grid1((size_t)m * c), TPB, 0, stream>>>(yb, xs[i], m, c);
    k_knn<<<grid1(m), TPB, 0, stream>>>(psc[i], psc[i], m, m, idxs[i]);
    for (int b = 0; b < NBLK[i] - 1; ++b)
      pt_block(encBlk[i][b], psc[i], xs[i], xs[i], idxs[i], m, c);
  }

  // ===================== decoder head (global context) =====================
  treset();
  k_mean_rows<<<grid1(1024), TPB, 0, stream>>>(xs[4], meanb, 32, 1024);
  k_gemv_relu<<<grid1(1024), TPB, 0, stream>>>(meanb, decUp2[4], gvec, 1024, 1024);
  float* xcat = (float*)walloc((size_t)32 * 2048 * 4);
  k_bcast_concat<<<grid1((size_t)32 * 2048), TPB, 0, stream>>>(xs[4], gvec, xcat, 32, 1024);
  gemm(xcat, decUp1[4], decx[4], 32, 1024, 2048);
  bn(decx[4], decx[4], 32, 1024, nullptr, 1);
  pt_block(decBlk[4], psc[4], decx[4], decx[4], idxs[4], 32, 1024);

  // ===================== decoder up path =====================
  const float* xc = decx[4];
  for (int i = 3; i >= 0; --i) {
    const int n = NP[i], c = CHN[i];
    const int nco = NP[i+1], cco = CHN[i+1];
    treset();
    float* a = (float*)walloc((size_t)n * c * 4);
    gemm(xs[i], decUp1[i], a, n, c, c);
    bn(a, a, n, c, nullptr, 1);
    float* bco = (float*)walloc((size_t)nco * c * 4);
    gemm(xc, decUp2[i], bco, nco, c, cco);
    bn(bco, bco, nco, c, nullptr, 1);
    k_interp_add<<<grid1((size_t)n), TPB, 0, stream>>>(psc[i+1], psc[i], bco, a, decx[i],
                                                       nco, n, c);
    float* outBuf = (i == 0) ? (float*)d_out : decx[i];
    pt_block(decBlk[i], psc[i], decx[i], outBuf, idxs[i], n, c);
    xc = decx[i];
  }
}
